// llm_gcn_3_36421322670434
// MI455X (gfx1250) — matrix operand order repaired and checked by reading, not compiled, not run on hardware
//
#include <hip/hip_runtime.h>
#include <hip/hip_bf16.h>

typedef _Float16 half_t;
typedef _Float16 h16  __attribute__((ext_vector_type(16)));            // 32B aligned
typedef _Float16 h8v  __attribute__((ext_vector_type(8)));             // 16B aligned
typedef _Float16 h8a4 __attribute__((ext_vector_type(8), aligned(4))); // LDS, 4B aligned
typedef float    f8   __attribute__((ext_vector_type(8)));

#define NST 565
#define NB  16

#define SHUF16(lo, hi) \
  __builtin_shufflevector(lo, hi, 0, 1, 2, 3, 4, 5, 6, 7, 8, 9, 10, 11, 12, 13, 14, 15)

// ---------------------------------------------------------------------------
// Utility / packing kernels (one-time pre-pass, all operands L2-resident after)
// ---------------------------------------------------------------------------
__global__ void zf16(half_t* __restrict__ d, int n) {
  int i = blockIdx.x * blockDim.x + threadIdx.x;
  if (i < n) d[i] = (half_t)0;
}

// conv weights (O, CH, 125) -> (O, CH, 128) zero-padded taps
__global__ void pack_conv_w(const float* __restrict__ s, half_t* __restrict__ d, int n) {
  int i = blockIdx.x * blockDim.x + threadIdx.x;
  if (i >= n) return;
  int oc = i >> 7, t = i & 127;
  d[i] = (t < 125) ? (half_t)s[oc * 125 + t] : (half_t)0;
}

// B (K, N) row-major f32 -> Bt (N, Kpad) row-major f16, zero-padded K tail
__global__ void pack_bt(const float* __restrict__ s, half_t* __restrict__ d,
                        int K, int N, int Kpad) {
  long i = (long)blockIdx.x * blockDim.x + threadIdx.x;
  if (i >= (long)N * Kpad) return;
  int j = (int)(i / Kpad), k = (int)(i - (long)j * Kpad);
  d[i] = (k < K) ? (half_t)s[(long)k * N + j] : (half_t)0;
}

__global__ void copy_f32(const float* __restrict__ s, float* __restrict__ d, int n) {
  int i = blockIdx.x * blockDim.x + threadIdx.x;
  if (i < n) d[i] = s[i];
}

__global__ void mv_pack(const float* __restrict__ s, half_t* __restrict__ d) {
  int i = blockIdx.x * blockDim.x + threadIdx.x;
  if (i >= NB * 576) return;
  int b = i / 576, j = i - b * 576;
  d[i] = (j < NST) ? (half_t)s[b * NST + j] : (half_t)0;
}

__global__ void meta_kernel(const float* __restrict__ meta, half_t* __restrict__ z) {
  int i = threadIdx.x;
  if (i >= NB * 5) return;
  int b = i / 5, j = i - b * 5;
  z[b * 1856 + 1280 + j] = (half_t)meta[b * 5 + j];
}

// deg[c] = 1 + sum_r adj[r][c]; dinv = deg^-1/2
__global__ void deg_kernel(const float* __restrict__ adj, float* __restrict__ dinv) {
  int c = blockIdx.x * blockDim.x + threadIdx.x;
  if (c >= NST) return;
  float s = 1.0f;
  for (int r = 0; r < NST; ++r) s += adj[r * NST + c];
  dinv[c] = (s > 0.f) ? rsqrtf(s) : 0.f;
}

// norm stored as (565, 576) with zero-padded K tail (symmetric, so norm == normT)
__global__ void norm_kernel(const float* __restrict__ adj, const float* __restrict__ dinv,
                            half_t* __restrict__ nh) {
  int i = blockIdx.x * blockDim.x + threadIdx.x;
  if (i >= NST * 576) return;
  int r = i / 576, c = i - r * 576;
  float v = 0.f;
  if (c < NST) v = dinv[r] * (adj[r * NST + c] + (r == c ? 1.f : 0.f)) * dinv[c];
  nh[i] = (half_t)v;
}

// ---------------------------------------------------------------------------
// Fused conv1+conv2+conv3 per (batch, station); intermediates in LDS.
// K padded to 128 per input channel so each 32-wide k-tile lives in one
// channel: ch = k0>>7 (scalar), t = k&127 (contiguous) -> vector LDS loads.
//   conv1: M=438, N=32,  Kpad=384   conv2: M=157, N=64, Kpad=4096
//   conv3: M=17,  N=128, Kpad=8192
// ---------------------------------------------------------------------------
__global__ __launch_bounds__(128) void conv_fused(
    const float* __restrict__ wav,
    const half_t* __restrict__ w1, const float* __restrict__ bc1,
    const half_t* __restrict__ w2, const float* __restrict__ bc2,
    const half_t* __restrict__ w3, const float* __restrict__ bc3,
    half_t* __restrict__ feat) {
  __shared__ half_t sIn[3 * 1024];
  __shared__ half_t sC1[32 * 448];
  __shared__ half_t sC2[64 * 192];

  int blk = blockIdx.x;
  int b = blk / NST, n = blk - b * NST;
  int tid = threadIdx.x, wave = tid >> 5, lane = tid & 31;
  int laneM = lane & 15;
  int kb8 = (lane >> 4) << 3;   // accumulator row sub-offset (D row 8*(lane/16))
  int kb16 = (lane >> 4) << 4;  // B-fragment k sub-offset

  const float* src = wav + ((long)(b * NST + n)) * 3000;
  for (int i = tid; i < 3072; i += 128) {
    int ch = i >> 10, t = i & 1023;
    sIn[i] = (t < 1000) ? (half_t)src[t * 3 + ch] : (half_t)0;
  }
  __syncthreads();

  // ---- phase 1: conv1 -> sC1[ch][pos] (rows 0..447, zeros past 437) ----
  for (int job = wave; job < 56; job += 4) {     // 28 m-tiles x 2 n-tiles
    int m0 = (job >> 1) * 16, n0 = (job & 1) * 16;
    int row = m0 + laneM, col = n0 + laneM;
    const half_t* pb = w1 + col * 384 + kb16;
    f8 acc = {};
    for (int k0 = 0; k0 < 384; k0 += 32) {
      const half_t* pa = sIn + ((k0 >> 7) << 10) + 2 * row + (k0 & 127) + kb16;
      h8a4 lo = *(const h8a4*)pa;
      h8a4 hi = *(const h8a4*)(pa + 8);
      h16 a = SHUF16(lo, hi);
      h16 bf = *(const h16*)(pb + k0);
      acc = __builtin_amdgcn_wmma_f32_16x16x32_f16(false, a, false, bf, (short)0, acc,
                                                   false, false);
    }
    float bias = bc1[col];
#pragma unroll
    for (int r = 0; r < 8; ++r) {
      int orow = m0 + r + kb8;
      sC1[col * 448 + orow] =
          (orow < 438) ? (half_t)fmaxf(acc[r] + bias, 0.f) : (half_t)0;
    }
  }
  __syncthreads();

  // ---- phase 2: conv2 -> sC2[ch][pos] (rows 0..159, zeros past 156) ----
  for (int job = wave; job < 40; job += 4) {     // 10 m-tiles x 4 n-tiles
    int m0 = (job >> 2) * 16, n0 = (job & 3) * 16;
    int row = m0 + laneM, col = n0 + laneM;
    const half_t* pb = w2 + col * 4096 + kb16;
    f8 acc = {};
    for (int k0 = 0; k0 < 4096; k0 += 32) {
      const half_t* pa = sC1 + (k0 >> 7) * 448 + 2 * row + (k0 & 127) + kb16;
      h8a4 lo = *(const h8a4*)pa;
      h8a4 hi = *(const h8a4*)(pa + 8);
      h16 a = SHUF16(lo, hi);
      h16 bf = *(const h16*)(pb + k0);
      acc = __builtin_amdgcn_wmma_f32_16x16x32_f16(false, a, false, bf, (short)0, acc,
                                                   false, false);
    }
    float bias = bc2[col];
#pragma unroll
    for (int r = 0; r < 8; ++r) {
      int orow = m0 + r + kb8;
      sC2[col * 192 + orow] =
          (orow < 157) ? (half_t)fmaxf(acc[r] + bias, 0.f) : (half_t)0;
    }
  }
  // zero sC2 tail positions 160..191 (read only by discarded rows, keep finite)
  for (int i = tid; i < 64 * 32; i += 128) sC2[(i >> 5) * 192 + 160 + (i & 31)] = (half_t)0;
  __syncthreads();

  // ---- phase 3: conv3 -> feat (global f16, torch raw-reshape layout) ----
  for (int job = wave; job < 16; job += 4) {     // 2 m-tiles x 8 n-tiles
    int m0 = (job >> 3) * 16, n0 = (job & 7) * 16;
    int row = m0 + laneM, col = n0 + laneM;
    const half_t* pb = w3 + col * 8192 + kb16;
    f8 acc = {};
    for (int k0 = 0; k0 < 8192; k0 += 32) {
      const half_t* pa = sC2 + (k0 >> 7) * 192 + 2 * row + (k0 & 127) + kb16;
      h8a4 lo = *(const h8a4*)pa;
      h8a4 hi = *(const h8a4*)(pa + 8);
      h16 a = SHUF16(lo, hi);
      h16 bf = *(const h16*)(pb + k0);
      acc = __builtin_amdgcn_wmma_f32_16x16x32_f16(false, a, false, bf, (short)0, acc,
                                                   false, false);
    }
    float bias = bc3[col];
#pragma unroll
    for (int r = 0; r < 8; ++r) {
      int orow = m0 + r + kb8;   // output width 0..16
      if (orow < 17)
        feat[(long)b * 1229440 + (long)col * 9605 + n * 17 + orow] =
            (half_t)fmaxf(acc[r] + bias, 0.f);
    }
  }
}

// ---------------------------------------------------------------------------
// Generic WMMA GEMM: C = A (MxKpad, row-major f16) x Bt^T (Bt: NxKpad f16).
// All K reads unguarded (operands zero-padded); row/col clamped (discarded
// rows/cols absorb the garbage). EPI: 0 none, 1 relu, 2 tanh.
// OUTM 0: outH[z*strideO + row*ldo + colOff + col]        (f16)
// OUTM 1: heads f32: outF[(col/565)*9040 + row*565 + col%565]
// OUTM 2: transposed batch layout: outH[col*ldo + (row/565)*576 + row%565]
// OUTM 3: aggregate: outH[((z*565 + col)*64 + row)]
// BROW: bias indexed by row (1) or col (0).
// ---------------------------------------------------------------------------
template <int EPI, int OUTM, int BROW>
__global__ __launch_bounds__(128) void gemm_wmma(
    const half_t* __restrict__ A, int lda, long strideA,
    const half_t* __restrict__ Bt, int ldk, long strideB,
    const float* __restrict__ bias, int M, int N, int Kpad,
    half_t* __restrict__ outH, int ldo, int colOff, long strideO,
    float* __restrict__ outF) {
  int wave = threadIdx.x >> 5, lane = threadIdx.x & 31;
  int nt = blockIdx.y * 4 + wave;
  int n0 = nt * 16, m0 = blockIdx.x * 16;
  if (n0 >= N) return;
  int z = blockIdx.z;
  A += (long)z * strideA;
  Bt += (long)z * strideB;

  int row = m0 + (lane & 15); if (row >= M) row = M - 1;
  int col = n0 + (lane & 15); if (col >= N) col = N - 1;
  const half_t* pa = A + (long)row * lda + ((lane >> 4) << 4);
  const half_t* pb = Bt + (long)col * ldk + ((lane >> 4) << 4);

  f8 acc = {};
  for (int k0 = 0; k0 < Kpad; k0 += 32) {
    h8v lo = *(const h8v*)(pa + k0);
    h8v hi = *(const h8v*)(pa + k0 + 8);
    h16 a = SHUF16(lo, hi);
    h16 b = *(const h16*)(pb + k0);
    acc = __builtin_amdgcn_wmma_f32_16x16x32_f16(false, a, false, b, (short)0, acc,
                                                 false, false);
  }

  int ocol = n0 + (lane & 15);
  if (ocol >= N) return;
  float bvc = (!BROW && bias) ? bias[ocol] : 0.f;
#pragma unroll
  for (int r = 0; r < 8; ++r) {
    int orow = m0 + r + ((lane >> 4) << 3);
    if (orow >= M) continue;
    float v = acc[r] + (BROW ? bias[orow] : bvc);
    if (EPI == 1) v = fmaxf(v, 0.f);
    if (EPI == 2) v = tanhf(v);
    if (OUTM == 0) {
      outH[(long)z * strideO + (long)orow * ldo + colOff + ocol] = (half_t)v;
    } else if (OUTM == 1) {
      int hs = ocol / NST, j = ocol - hs * NST;
      outF[(long)hs * (NB * NST) + (long)orow * NST + j] = v;
    } else if (OUTM == 2) {
      int bb = orow / NST, nn = orow - bb * NST;
      outH[(long)ocol * ldo + bb * 576 + nn] = (half_t)v;
    } else {
      outH[((long)z * NST + ocol) * 64 + orow] = (half_t)v;
    }
  }
}

// ---------------------------------------------------------------------------
extern "C" void kernel_launch(void* const* d_in, const int* in_sizes, int n_in,
                              void* d_out, int out_size, void* d_ws, size_t ws_size,
                              hipStream_t stream) {
  const float* wav  = (const float*)d_in[0];
  const float* adj  = (const float*)d_in[2];
  const float* meta = (const float*)d_in[3];
  const float* mv   = (const float*)d_in[4];
  const float* w_conv1 = (const float*)d_in[5];  const float* b_conv1 = (const float*)d_in[6];
  const float* w_conv2 = (const float*)d_in[7];  const float* b_conv2 = (const float*)d_in[8];
  const float* w_conv3 = (const float*)d_in[9];  const float* b_conv3 = (const float*)d_in[10];
  const float* w_gcn1 = (const float*)d_in[11];  const float* b_gcn1 = (const float*)d_in[12];
  const float* w_gcn2 = (const float*)d_in[13];  const float* b_gcn2 = (const float*)d_in[14];
  const float* w_fce  = (const float*)d_in[15];  const float* b_fce  = (const float*)d_in[16];
  const float* w_fc1  = (const float*)d_in[17];  const float* b_fc1  = (const float*)d_in[18];
  const float* w_head[5] = {(const float*)d_in[19], (const float*)d_in[21],
                            (const float*)d_in[23], (const float*)d_in[25],
                            (const float*)d_in[27]};
  const float* b_head[5] = {(const float*)d_in[20], (const float*)d_in[22],
                            (const float*)d_in[24], (const float*)d_in[26],
                            (const float*)d_in[28]};
  float* out = (float*)d_out;

  size_t off = 0;
  auto alloc = [&](size_t bytes) -> void* {
    void* p = (char*)d_ws + off;
    off += (bytes + 255) & ~(size_t)255;
    return p;
  };
  half_t* wc1p  = (half_t*)alloc(12288u * 2);        // (32, 384)
  half_t* wc2p  = (half_t*)alloc(262144u * 2);       // (64, 4096)
  half_t* wc3p  = (half_t*)alloc(1048576u * 2);      // (128, 8192)
  half_t* wg1t  = (half_t*)alloc(139264u * 2);       // (64, 2176)
  half_t* wg2t  = (half_t*)alloc(4096u * 2);         // (64, 64)
  half_t* wfet  = (half_t*)alloc(325440u * 2);       // (565, 576)
  half_t* wfc1t = (half_t*)alloc(46284800ull * 2);   // (1280, 36160)
  half_t* whht  = (half_t*)alloc(5243200ull * 2);    // (2825, 1856)
  float*  bhh   = (float*) alloc(2825u * 4);
  float*  dinv  = (float*) alloc(NST * 4);
  half_t* normh = (half_t*)alloc(325440u * 2);       // (565, 576) zero-padded
  half_t* feat  = (half_t*)alloc(19671040ull * 2);   // (16, 565*2176)
  half_t* h1t   = (half_t*)alloc(589824u * 2);       // (64, 16, 576)
  half_t* h2t   = (half_t*)alloc(589824u * 2);
  half_t* x1    = (half_t*)alloc(578560u * 2);       // (9040, 64)
  half_t* x2    = (half_t*)alloc(578560u * 2);
  half_t* mvh   = (half_t*)alloc(9216u * 2);         // (16, 576)
  half_t* zh    = (half_t*)alloc(29696u * 2);        // (16, 1856)
  if (off > ws_size) return;

  dim3 blk128(128), blk256(256);
  auto g1 = [](int n) { return dim3((n + 255) / 256); };

  // --- pre-pass: pack weights (f16, transposed/padded) ---
  pack_conv_w<<<g1(12288), blk256, 0, stream>>>(w_conv1, wc1p, 12288);
  pack_conv_w<<<g1(262144), blk256, 0, stream>>>(w_conv2, wc2p, 262144);
  pack_conv_w<<<g1(1048576), blk256, 0, stream>>>(w_conv3, wc3p, 1048576);
  pack_bt<<<g1(139264), blk256, 0, stream>>>(w_gcn1, wg1t, 2176, 64, 2176);
  pack_bt<<<g1(4096), blk256, 0, stream>>>(w_gcn2, wg2t, 64, 64, 64);
  pack_bt<<<g1(325440), blk256, 0, stream>>>(w_fce, wfet, NST, NST, 576);
  pack_bt<<<g1(46284800), blk256, 0, stream>>>(w_fc1, wfc1t, 36160, 1280, 36160);
  for (int h = 0; h < 5; ++h) {
    pack_bt<<<g1(1048640), blk256, 0, stream>>>(w_head[h], whht + (long)h * NST * 1856,
                                                1850, NST, 1856);
    copy_f32<<<dim3(3), blk256, 0, stream>>>(b_head[h], bhh + h * NST, NST);
  }
  mv_pack<<<g1(NB * 576), blk256, 0, stream>>>(mv, mvh);

  // --- GCN normalization matrix ---
  deg_kernel<<<dim3(5), blk128, 0, stream>>>(adj, dinv);
  norm_kernel<<<g1(NST * 576), blk256, 0, stream>>>(adj, dinv, normh);

  // --- fused conv stack: one workgroup per (b, station) ---
  conv_fused<<<dim3(NB * NST), blk128, 0, stream>>>(wav, wc1p, b_conv1, wc2p, b_conv2,
                                                    wc3p, b_conv3, feat);

  // --- GCN layer 1: h1^T = (feat @ Wg1)^T ; x1 = relu(h1^T @ norm + b)^T ---
  zf16<<<g1(589824), blk256, 0, stream>>>(h1t, 589824);
  gemm_wmma<0, 2, 0><<<dim3(565, 1, 1), blk128, 0, stream>>>(
      feat, 2176, 0, wg1t, 2176, 0, nullptr, 9040, 64, 2176, h1t, 9216, 0, 0, nullptr);
  gemm_wmma<1, 3, 1><<<dim3(4, 9, NB), blk128, 0, stream>>>(
      h1t, 9216, 576, normh, 576, 0, b_gcn1, 64, NST, 576, x1, 0, 0, 0, nullptr);

  // --- GCN layer 2: h2^T = (x1 @ Wg2)^T ; x2 = tanh(h2^T @ norm + b)^T ---
  zf16<<<g1(589824), blk256, 0, stream>>>(h2t, 589824);
  gemm_wmma<0, 2, 0><<<dim3(565, 1, 1), blk128, 0, stream>>>(
      x1, 64, 0, wg2t, 64, 0, nullptr, 9040, 64, 64, h2t, 9216, 0, 0, nullptr);
  gemm_wmma<2, 3, 1><<<dim3(4, 9, NB), blk128, 0, stream>>>(
      h2t, 9216, 576, normh, 576, 0, b_gcn2, 64, NST, 576, x2, 0, 0, 0, nullptr);

  // --- z: zero, then fc1 / meta / fc_extra ---
  zf16<<<g1(29696), blk256, 0, stream>>>(zh, 29696);
  gemm_wmma<1, 0, 0><<<dim3(1, 20, 1), blk128, 0, stream>>>(
      x2, 36160, 0, wfc1t, 36160, 0, b_fc1, NB, 1280, 36160, zh, 1856, 0, 0, nullptr);
  gemm_wmma<1, 0, 0><<<dim3(1, 9, 1), blk128, 0, stream>>>(
      mvh, 576, 0, wfet, 576, 0, b_fce, NB, NST, 576, zh, 1856, 1285, 0, nullptr);
  meta_kernel<<<dim3(1), blk128, 0, stream>>>(meta, zh);

  // --- 5 heads fused: (16, 1856) @ (1856, 2825) -> d_out (f32) ---
  gemm_wmma<0, 1, 0><<<dim3(1, 45, 1), blk128, 0, stream>>>(
      zh, 1856, 0, whht, 1856, 0, bhh, NB, 2825, 1856, nullptr, 0, 0, 0, out);
}
